// SingleBlock_343597383847
// MI455X (gfx1250) — matrix operand order repaired and checked by reading, not compiled, not run on hardware
//
#include <hip/hip_runtime.h>
#include <hip/hip_bf16.h>
#include <math.h>

typedef unsigned short u16t;
typedef unsigned int   u32t;
typedef __attribute__((ext_vector_type(16))) __bf16 v16bf;
typedef __attribute__((ext_vector_type(8)))  float  v8f;

#define NUM_HEADS 8
#define DH 128
#define DMODEL 1024
#define TSTRIDE 40          // padded u16 row stride for transposed LDS tiles (80B, 16B-aligned)

__device__ __forceinline__ u16t f2bf(float f) {
  union { float f; u32t u; } x; x.f = f;
  u32t r = x.u + 0x7fffu + ((x.u >> 16) & 1u);
  return (u16t)(r >> 16);
}

union FragBF { v16bf v; u32t u[8]; };

// 16x32 bf16 A fragment from row-major source (ld elements per row).
// ISA layout: lane m=lane&15 (row), half=lane>>4; VGPR j<4 -> K=2j+8*half,
// VGPR j>=4 -> K=16+2(j-4)+8*half; packed pairs contiguous -> b128-mergeable.
__device__ __forceinline__ void load_frag_a(FragBF& f, const u16t* base, int ld, int lane) {
  int m = lane & 15, hf = lane >> 4;
  const u16t* p = base + (size_t)m * ld;
#pragma unroll
  for (int j = 0; j < 8; ++j) {
    int k0 = (j < 4) ? (2 * j + 8 * hf) : (16 + 2 * (j - 4) + 8 * hf);
    f.u[j] = *(const u32t*)(p + k0);
  }
}

// 32x16 B fragment where source stores K contiguously per column:
// element (k,n) = base[n*ld + k].  ISA layout: lane n=lane&15, half=lane>>4;
// VGPR j<4 -> K=2j+8*half, VGPR j>=4 -> K=16+2(j-4)+8*half (matches A fragment).
__device__ __forceinline__ void load_frag_b_kcontig(FragBF& f, const u16t* base, int ld, int lane) {
  int n = lane & 15, hf = lane >> 4;
  const u16t* p = base + (size_t)n * ld;
#pragma unroll
  for (int j = 0; j < 8; ++j) {
    int k0 = (j < 4) ? (2 * j + 8 * hf) : (16 + 2 * (j - 4) + 8 * hf);
    f.u[j] = *(const u32t*)(p + k0);
  }
}

// ---------------- GEMM: C = A_bf16[MxK] @ B_bf16[KxN] + bias, fused epilogue ---
// Block tile 64x128, 4 waves (each 16 rows x 128 cols), BK=32.
// mode 0: write fp32 to outF.  mode 1: QKV epilogue -> mask*(1+gate), split K/Q/V,
// fold 1/sqrt(dh) into Q, write bf16 into attention-layout buffers.
__global__ __launch_bounds__(128)
void gemm_bf16_kernel(const u16t* __restrict__ A, const u16t* __restrict__ Bw,
                      const float* __restrict__ bias,
                      int M, int N, int K, int mode,
                      float* __restrict__ outF,
                      u16t* __restrict__ Kb, u16t* __restrict__ Qb, u16t* __restrict__ Vb,
                      const float* __restrict__ mask, const float* __restrict__ gate,
                      int S, int kld, int koff, float qscale)
{
  __shared__ __align__(16) u16t As[64 * 32];            // [row][k] row-major
  __shared__ __align__(16) u16t BsT[128 * TSTRIDE];     // [n][k] transposed, padded
  const int lane = threadIdx.x & 31;
  const int wave = threadIdx.x >> 5;
  const int bm = blockIdx.y * 64;
  const int bn = blockIdx.x * 128;

  v8f acc[8] = {};

  const int nk = K >> 5;
  for (int kt = 0; kt < nk; ++kt) {
    const u32t* Ap = (const u32t*)(A + (size_t)bm * K + kt * 32);
    const u32t* Bp = (const u32t*)(Bw + (size_t)(kt * 32) * N + bn);
    u32t* As32 = (u32t*)As;
#pragma unroll
    for (int t = 0; t < 8; ++t) {                       // A: 64x32 = 1024 dwords
      int idx = threadIdx.x + t * 128;
      As32[idx] = Ap[(size_t)(idx >> 4) * (K >> 1) + (idx & 15)];
    }
#pragma unroll
    for (int t = 0; t < 16; ++t) {                      // B: 32x128 = 2048 dwords, transpose
      int idx = threadIdx.x + t * 128;
      int kk = idx >> 6;                                // k row 0..31
      int nn = (idx & 63) * 2;                          // col pair
      u32t d = Bp[(size_t)kk * (N >> 1) + (idx & 63)];
      BsT[(size_t)nn * TSTRIDE + kk]       = (u16t)d;
      BsT[(size_t)(nn + 1) * TSTRIDE + kk] = (u16t)(d >> 16);
    }
    __syncthreads();

    // issue all fragment loads before the first WMMA so ds latency overlaps math
    FragBF fa; load_frag_a(fa, As + wave * 16 * 32, 32, lane);
    FragBF fb[8];
#pragma unroll
    for (int nt = 0; nt < 8; ++nt)
      load_frag_b_kcontig(fb[nt], BsT + (size_t)(nt * 16) * TSTRIDE, TSTRIDE, lane);
#pragma unroll
    for (int nt = 0; nt < 8; ++nt)
      acc[nt] = __builtin_amdgcn_wmma_f32_16x16x32_bf16(
          false, fa.v, false, fb[nt].v, (short)0, acc[nt], false, false);
    __syncthreads();
  }

  const int n0 = lane & 15, hf = lane >> 4;
#pragma unroll
  for (int nt = 0; nt < 8; ++nt) {
#pragma unroll
    for (int r = 0; r < 8; ++r) {
      int row = bm + wave * 16 + r + 8 * hf;
      int col = bn + nt * 16 + n0;
      float y = acc[nt][r] + bias[col];
      if (mode == 0) {
        outF[(size_t)row * N + col] = y;
      } else {
        int b = row / S, s = row - b * S;
        int region = col >> 10;        // D = 1024
        int jd = col & 1023;
        y *= mask[(size_t)b * S + s];
        if (region == 0) {
          if (gate) y *= (1.0f + gate[(size_t)b * DMODEL + jd]);
          Kb[((size_t)b * kld + s + koff) * DMODEL + jd] = f2bf(y);
        } else if (region == 1) {
          if (gate) y *= (1.0f + gate[(size_t)b * DMODEL + jd]);
          Qb[((size_t)b * S + s) * DMODEL + jd] = f2bf(y * qscale);
        } else {
          Vb[((size_t)b * kld + s + koff) * DMODEL + jd] = f2bf(y);
        }
      }
    }
  }
}

// ---------------- Flash attention: 4 waves/block, 16 query rows per wave -------
__global__ __launch_bounds__(128)
void attn_kernel(const u16t* __restrict__ Qb, const u16t* __restrict__ Kb,
                 const u16t* __restrict__ Vb, const float* __restrict__ maskm,
                 const float* __restrict__ Xres, u16t* __restrict__ Ub,
                 int S, int Tpad, int kld)
{
  __shared__ __align__(16) u16t Ks[32 * DH];            // [t][d] (d = contraction for QK^T)
  __shared__ __align__(16) u16t VsT[DH * TSTRIDE];      // [d][t] transposed, padded
  __shared__ __align__(16) u16t Ps[4][16 * 32];

  const int lane = threadIdx.x & 31;
  const int wave = threadIdx.x >> 5;
  const int b = blockIdx.z;
  const int h = blockIdx.y;
  const int s0 = blockIdx.x * 64 + wave * 16;
  const int n0 = lane & 15, hf = lane >> 4;

  FragBF qf[4];
  {
    const u16t* qbase = Qb + ((size_t)b * S + s0) * DMODEL + h * DH;
#pragma unroll
    for (int ks = 0; ks < 4; ++ks)
      load_frag_a(qf[ks], qbase + ks * 32, DMODEL, lane);
  }

  v8f o[8] = {};
  float mi[8], li[8];
#pragma unroll
  for (int r = 0; r < 8; ++r) { mi[r] = -3.0e38f; li[r] = 0.0f; }

  for (int t0 = 0; t0 < Tpad; t0 += 32) {
    {
      const u32t* Kp = (const u32t*)(Kb + ((size_t)b * kld + t0) * DMODEL + h * DH);
      const u32t* Vp = (const u32t*)(Vb + ((size_t)b * kld + t0) * DMODEL + h * DH);
      u32t* Ks32 = (u32t*)Ks;
#pragma unroll
      for (int t = 0; t < 16; ++t) {                    // K: 32x128 = 2048 dwords
        int idx = threadIdx.x + t * 128;
        Ks32[idx] = Kp[(size_t)(idx >> 6) * (DMODEL >> 1) + (idx & 63)];
      }
#pragma unroll
      for (int t = 0; t < 16; ++t) {                    // V: transpose to [d][t]
        int idx = threadIdx.x + t * 128;
        int tt = idx >> 6;                              // key row 0..31
        int dd = (idx & 63) * 2;                        // d pair
        u32t dv = Vp[(size_t)tt * (DMODEL >> 1) + (idx & 63)];
        VsT[(size_t)dd * TSTRIDE + tt]       = (u16t)dv;
        VsT[(size_t)(dd + 1) * TSTRIDE + tt] = (u16t)(dv >> 16);
      }
    }
    __syncthreads();

    // S = Q (16x128) x K^T (128x32) as two 16-col tiles, K-loop of 4
    v8f sc[2] = {};
#pragma unroll
    for (int ks = 0; ks < 4; ++ks) {
      FragBF kf0, kf1;                                  // preload both tiles, then 2 WMMAs
      load_frag_b_kcontig(kf0, Ks + (size_t)0 * DH + ks * 32, DH, lane);
      load_frag_b_kcontig(kf1, Ks + (size_t)16 * DH + ks * 32, DH, lane);
      sc[0] = __builtin_amdgcn_wmma_f32_16x16x32_bf16(
          false, qf[ks].v, false, kf0.v, (short)0, sc[0], false, false);
      sc[1] = __builtin_amdgcn_wmma_f32_16x16x32_bf16(
          false, qf[ks].v, false, kf1.v, (short)0, sc[1], false, false);
    }

    float mv0 = maskm[(size_t)b * Tpad + t0 + n0];
    float mv1 = maskm[(size_t)b * Tpad + t0 + 16 + n0];
#pragma unroll
    for (int r = 0; r < 8; ++r) {
      if (mv0 == 0.0f) sc[0][r] = -1.0e9f;
      if (mv1 == 0.0f) sc[1][r] = -1.0e9f;
    }

    // streaming softmax (row reductions across 16-lane halves)
#pragma unroll
    for (int r = 0; r < 8; ++r) {
      float mx = fmaxf(sc[0][r], sc[1][r]);
#pragma unroll
      for (int d = 1; d < 16; d <<= 1) mx = fmaxf(mx, __shfl_xor(mx, d, 32));
      float mnew = fmaxf(mi[r], mx);
      float corr = __expf(mi[r] - mnew);
      mi[r] = mnew;
      float p0 = __expf(sc[0][r] - mnew);
      float p1 = __expf(sc[1][r] - mnew);
      sc[0][r] = p0; sc[1][r] = p1;
      float ps = p0 + p1;
#pragma unroll
      for (int d = 1; d < 16; d <<= 1) ps += __shfl_xor(ps, d, 32);
      li[r] = li[r] * corr + ps;
#pragma unroll
      for (int dt = 0; dt < 8; ++dt) o[dt][r] *= corr;
    }

    // P tile (16x32) to LDS for re-layout into an A fragment
    {
      u16t* pp = Ps[wave];
#pragma unroll
      for (int r = 0; r < 8; ++r) {
        pp[(r + 8 * hf) * 32 + n0]      = f2bf(sc[0][r]);
        pp[(r + 8 * hf) * 32 + 16 + n0] = f2bf(sc[1][r]);
      }
    }
    __syncthreads();

    // O += P (16x32) x V (32x128), V fragments preloaded in batches of 4
    FragBF pf; load_frag_a(pf, Ps[wave], 32, lane);
#pragma unroll
    for (int g = 0; g < 2; ++g) {
      FragBF vf[4];
#pragma unroll
      for (int j = 0; j < 4; ++j)
        load_frag_b_kcontig(vf[j], VsT + (size_t)((g * 4 + j) * 16) * TSTRIDE, TSTRIDE, lane);
#pragma unroll
      for (int j = 0; j < 4; ++j)
        o[g * 4 + j] = __builtin_amdgcn_wmma_f32_16x16x32_bf16(
            false, pf.v, false, vf[j].v, (short)0, o[g * 4 + j], false, false);
    }
    __syncthreads();
  }

  // normalize + residual, store bf16 for the output projection
#pragma unroll
  for (int dt = 0; dt < 8; ++dt) {
#pragma unroll
    for (int r = 0; r < 8; ++r) {
      int s = s0 + r + 8 * hf;
      int d = h * DH + dt * 16 + n0;
      float val = o[dt][r] / li[r];
      float res = Xres[((size_t)b * S + s) * DMODEL + d];
      Ub[((size_t)b * S + s) * DMODEL + d] = f2bf(res + val);
    }
  }
}

// ---------------- small fp32 kernels -------------------------------------------
__global__ void masked_mean_kernel(const float* __restrict__ X, const float* __restrict__ mask,
                                   float* __restrict__ mean, int S) {
  int b = blockIdx.y;
  int d = blockIdx.x * blockDim.x + threadIdx.x;
  float acc = 0.f, msum = 0.f;
  for (int s = 0; s < S; ++s) {
    float mv = mask[(size_t)b * S + s];
    acc += X[((size_t)b * S + s) * DMODEL + d] * mv;
    msum += mv;
  }
  mean[(size_t)b * DMODEL + d] = acc / msum;
}

__global__ void gate_kernel(const float* __restrict__ mean, const float* __restrict__ W,
                            const float* __restrict__ bias, float* __restrict__ gate) {
  int b = blockIdx.y;
  int j = blockIdx.x * blockDim.x + threadIdx.x;
  float acc = bias[j];
  for (int d = 0; d < DMODEL; ++d) {
    float x = fmaxf(mean[(size_t)b * DMODEL + d], 0.f);
    acc += x * W[(size_t)d * DMODEL + j];
  }
  gate[(size_t)b * DMODEL + j] = 1.0f / (1.0f + __expf(-acc));
}

__global__ void relu_bf16_kernel(const float* __restrict__ X, u16t* __restrict__ Y, size_t n) {
  size_t i = (size_t)blockIdx.x * blockDim.x + threadIdx.x;
  if (i < n) Y[i] = f2bf(fmaxf(X[i], 0.f));
}

__global__ void cvt_bf16_kernel(const float* __restrict__ X, u16t* __restrict__ Y, size_t n) {
  size_t i = (size_t)blockIdx.x * blockDim.x + threadIdx.x;
  if (i < n) Y[i] = f2bf(X[i]);
}

__global__ void token_kernel(const float* __restrict__ mk, const float* __restrict__ mv,
                             const float* __restrict__ gate, u16t* __restrict__ Kb,
                             u16t* __restrict__ Vb, int kld, float sqrtD) {
  int b = blockIdx.y;
  int d = blockIdx.x * blockDim.x + threadIdx.x;
  float g = gate ? (1.0f + gate[(size_t)b * DMODEL + d]) : 1.0f;
  Kb[(size_t)b * kld * DMODEL + d] = f2bf(sqrtD * mk[d] * g);
  Vb[(size_t)b * kld * DMODEL + d] = f2bf(mv[d]);
}

__global__ void maskm_kernel(const float* __restrict__ mask, float* __restrict__ maskm,
                             int S, int Ttok, int Tpad, int koff) {
  int b = blockIdx.x;
  for (int t = threadIdx.x; t < Tpad; t += blockDim.x) {
    float v;
    if (koff == 1) {
      if (t == 0) v = 1.0f;
      else if (t < Ttok) v = mask[(size_t)b * S + (t - 1)];
      else v = 0.0f;
    } else {
      v = (t < S) ? mask[(size_t)b * S + t] : 0.0f;
    }
    maskm[(size_t)b * Tpad + t] = v;
  }
}

// ---------------- host launcher -------------------------------------------------
extern "C" void kernel_launch(void* const* d_in, const int* in_sizes, int n_in,
                              void* d_out, int out_size, void* d_ws, size_t ws_size,
                              hipStream_t stream) {
  (void)in_sizes; (void)n_in; (void)out_size; (void)ws_size;
  const int B = 32, D = DMODEL;
  const float* v      = (const float*)d_in[0];
  const float* q      = (const float*)d_in[1];
  const float* c      = (const float*)d_in[2];
  const float* v_mask = (const float*)d_in[3];
  const float* q_mask = (const float*)d_in[4];
  const float* c_mask = (const float*)d_in[5];
  const float* Wv4q = (const float*)d_in[6];  const float* bv4q = (const float*)d_in[7];
  const float* Wq4v = (const float*)d_in[8];  const float* bq4v = (const float*)d_in[9];

  char* ws = (char*)d_ws;
  size_t off = 0;
  auto alloc = [&](size_t bytes) -> void* {
    void* p = ws + off;
    off += (bytes + 255) & ~(size_t)255;
    return p;
  };

  float* meanV   = (float*)alloc((size_t)B * D * 4);
  float* meanQ   = (float*)alloc((size_t)B * D * 4);
  float* gateV4Q = (float*)alloc((size_t)B * D * 4);   // from v_mean, applied to q stream
  float* gateQ4V = (float*)alloc((size_t)B * D * 4);   // from q_mean, applied to v stream
  u16t*  Xb  = (u16t*)alloc((size_t)B * 512 * D * 2);  // relu(x) bf16; later aliased as Ub
  u16t*  Wb  = (u16t*)alloc((size_t)D * 3 * D * 2);
  u16t*  Wob = (u16t*)alloc((size_t)D * D * 2);
  u16t*  Kb  = (u16t*)alloc((size_t)B * 544 * D * 2);
  u16t*  Vbuf= (u16t*)alloc((size_t)B * 544 * D * 2);
  u16t*  Qb  = (u16t*)alloc((size_t)B * 512 * D * 2);
  float* maskm = (float*)alloc((size_t)B * 544 * 4);
  u16t*  Ub  = Xb;  // safe alias: Xb dead after QKV GEMM, kernels serialized on stream

  masked_mean_kernel<<<dim3(D / 256, B), 256, 0, stream>>>(v, v_mask, meanV, 512);
  masked_mean_kernel<<<dim3(D / 256, B), 256, 0, stream>>>(q, q_mask, meanQ, 256);
  gate_kernel<<<dim3(D / 64, B), 64, 0, stream>>>(meanV, Wv4q, bv4q, gateV4Q);
  gate_kernel<<<dim3(D / 64, B), 64, 0, stream>>>(meanQ, Wq4v, bq4v, gateQ4V);

  struct StreamDesc {
    const float *X, *mask, *W, *bias, *gate, *mk, *mvv, *Wo, *bo;
    int S, Ttok, Tpad, koff; size_t outOff;
  };
  const StreamDesc sd[3] = {
    { v, v_mask, (const float*)d_in[10], (const float*)d_in[11], gateQ4V,
      (const float*)d_in[16], (const float*)d_in[17],
      (const float*)d_in[20], (const float*)d_in[21], 512, 513, 544, 1, 0 },
    { q, q_mask, (const float*)d_in[12], (const float*)d_in[13], gateV4Q,
      nullptr, nullptr,
      (const float*)d_in[22], (const float*)d_in[23], 256, 256, 256, 0, (size_t)B * 512 * D },
    { c, c_mask, (const float*)d_in[14], (const float*)d_in[15], nullptr,
      (const float*)d_in[18], (const float*)d_in[19],
      (const float*)d_in[24], (const float*)d_in[25], 512, 513, 544, 1, (size_t)B * (512 + 256) * D },
  };

  const float qscale = 0.08838834764831845f;   // 1/sqrt(dh=128)
  for (int i = 0; i < 3; ++i) {
    const StreamDesc& st = sd[i];
    const int M = B * st.S;
    const size_t nX = (size_t)M * D;

    relu_bf16_kernel<<<(unsigned)((nX + 255) / 256), 256, 0, stream>>>(st.X, Xb, nX);
    const size_t nW = (size_t)D * 3 * D;
    cvt_bf16_kernel<<<(unsigned)((nW + 255) / 256), 256, 0, stream>>>(st.W, Wb, nW);
    maskm_kernel<<<B, 256, 0, stream>>>(st.mask, maskm, st.S, st.Ttok, st.Tpad, st.koff);
    if (st.koff)
      token_kernel<<<dim3(D / 256, B), 256, 0, stream>>>(st.mk, st.mvv,
          (i == 0) ? st.gate : nullptr, Kb, Vbuf, st.Tpad, 32.0f /* sqrt(D) */);

    // QKV projection with fused mask/gate/split epilogue
    gemm_bf16_kernel<<<dim3(3 * D / 128, M / 64), 128, 0, stream>>>(
        Xb, Wb, st.bias, M, 3 * D, D, 1,
        nullptr, Kb, Qb, Vbuf, st.mask, st.gate, st.S, st.Tpad, st.koff, qscale);

    // attention + residual -> Ub (bf16)
    attn_kernel<<<dim3(st.S / 64, NUM_HEADS, B), 128, 0, stream>>>(
        Qb, Kb, Vbuf, maskm, st.X, Ub, st.S, st.Tpad, st.Tpad);

    // output projection -> fp32 slice of d_out
    const size_t nWo = (size_t)D * D;
    cvt_bf16_kernel<<<(unsigned)((nWo + 255) / 256), 256, 0, stream>>>(st.Wo, Wob, nWo);
    gemm_bf16_kernel<<<dim3(D / 128, M / 64), 128, 0, stream>>>(
        Ub, Wob, st.bo, M, D, D, 0,
        (float*)d_out + st.outOff, nullptr, nullptr, nullptr, nullptr, nullptr,
        st.S, 0, 0, 1.0f);
  }
}
